// SwitchSparseMoe_46883863003224
// MI455X (gfx1250) — matrix operand order repaired and checked by reading, not compiled, not run on hardware
//
#include <hip/hip_runtime.h>
#include <hip/hip_bf16.h>

// ---------------- problem constants (match reference shapes) ----------------
constexpr int BB  = 4;
constexpr int SS  = 2048;
constexpr int DD  = 1024;
constexpr int EE  = 8;
constexpr int FF  = 4096;
constexpr int CAP = 320;
constexpr int TT  = BB * CAP;          // 1280 rows per expert
constexpr int NTOK = BB * SS;          // 8192 tokens

// ---------------- types ----------------
typedef __bf16 bf16_t;
typedef bf16_t v16bf __attribute__((ext_vector_type(16)));
typedef bf16_t v8bf  __attribute__((ext_vector_type(8)));
typedef float  v8f   __attribute__((ext_vector_type(8)));

static __device__ __forceinline__ unsigned short f2bf_bits(float f) {
  unsigned u = __float_as_uint(f);
  unsigned r = u + 0x7fffu + ((u >> 16) & 1u);   // round-to-nearest-even
  return (unsigned short)(r >> 16);
}

static __device__ __forceinline__ unsigned long long pack4bf(float4 v) {
  return (unsigned long long)f2bf_bits(v.x) |
         ((unsigned long long)f2bf_bits(v.y) << 16) |
         ((unsigned long long)f2bf_bits(v.z) << 32) |
         ((unsigned long long)f2bf_bits(v.w) << 48);
}

// ============================================================================
// 0) Weight pre-conversion f32 -> bf16 (makes W L2-resident for the GEMMs)
// ============================================================================
__global__ __launch_bounds__(256) void cvt_bf16_kernel(
    const float* __restrict__ src, unsigned short* __restrict__ dst) {
  const size_t i = (size_t)blockIdx.x * 256 + threadIdx.x;   // one float4 each
  const float4 v = ((const float4*)src)[i];
  ((unsigned long long*)dst)[i] = pack4bf(v);
}

// ============================================================================
// 1) Router: logits = data . gate_w^T ; softmax-max prob ; argmax expert
// ============================================================================
__global__ __launch_bounds__(256) void router_kernel(
    const float* __restrict__ data, const float* __restrict__ gate_w,
    float* __restrict__ logits_out, int* __restrict__ eidx,
    float* __restrict__ prob) {
  __shared__ float gate[EE * DD];                 // 32 KB
  const int tid = threadIdx.x;
#pragma unroll
  for (int i = 0; i < (EE * DD) / (256 * 4); ++i)
    ((float4*)gate)[tid + 256 * i] = ((const float4*)gate_w)[tid + 256 * i];
  __syncthreads();

  const int lane = tid & 31;
  const int wave = tid >> 5;
  const int tok  = blockIdx.x * 8 + wave;

  float acc[EE];
#pragma unroll
  for (int e = 0; e < EE; ++e) acc[e] = 0.f;

  const float* dp = data + (size_t)tok * DD;
#pragma unroll 4
  for (int i = 0; i < DD / 32; ++i) {
    const int idx = i * 32 + lane;
    const float x = dp[idx];
#pragma unroll
    for (int e = 0; e < EE; ++e) acc[e] += x * gate[e * DD + idx];
  }
#pragma unroll
  for (int off = 16; off > 0; off >>= 1)
#pragma unroll
    for (int e = 0; e < EE; ++e) acc[e] += __shfl_down(acc[e], off, 32);

  if (lane == 0) {
    float m = acc[0]; int am = 0;
#pragma unroll
    for (int e = 1; e < EE; ++e)
      if (acc[e] > m) { m = acc[e]; am = e; }
    float sum = 0.f;
#pragma unroll
    for (int e = 0; e < EE; ++e) sum += __expf(acc[e] - m);
#pragma unroll
    for (int e = 0; e < EE; ++e) logits_out[(size_t)tok * EE + e] = acc[e];
    prob[tok] = 1.f / sum;                        // max softmax prob
    eidx[tok] = am;
  }
}

// ============================================================================
// 2) Capacity scan (wave32 ballot), reproduces cumsum(onehot) priorities
// ============================================================================
__global__ __launch_bounds__(32) void scan_kernel(
    const int* __restrict__ eidx, int* __restrict__ slotpos,
    int* __restrict__ expert_index_out) {
  const int b = blockIdx.x;
  const int lane = threadIdx.x;
  const unsigned lt = (1u << lane) - 1u;
  int counts[EE];
#pragma unroll
  for (int e = 0; e < EE; ++e) counts[e] = 0;

  for (int s0 = 0; s0 < SS; s0 += 32) {
    const int t = b * SS + s0 + lane;
    const int e = eidx[t];
    int pos = 0;
#pragma unroll
    for (int ex = 0; ex < EE; ++ex) {
      const unsigned bal = (unsigned)__ballot(e == ex);
      if (e == ex) pos = counts[ex] + __popc(bal & lt);
      counts[ex] += __popc(bal);                  // uniform across lanes
    }
    const bool kept = pos < CAP;
    slotpos[t] = kept ? pos : -1;
    expert_index_out[t] = kept ? e : 0;           // argmax of all-zero mask = 0
  }
}

// ============================================================================
// 3) Dispatch: kept tokens -> Xin (E, B*CAP, D) in bf16
// ============================================================================
__global__ __launch_bounds__(256) void dispatch_kernel(
    const float* __restrict__ data, const int* __restrict__ eidx,
    const int* __restrict__ slotpos, unsigned short* __restrict__ xin) {
  const int tok = blockIdx.x;
  const int pos = slotpos[tok];
  if (pos < 0) return;
  const int e = eidx[tok];
  const int b = tok / SS;
  const size_t row = (size_t)e * TT + (size_t)b * CAP + pos;
  const float4 v = ((const float4*)(data + (size_t)tok * DD))[threadIdx.x];
  ((unsigned long long*)(xin + row * DD))[threadIdx.x] = pack4bf(v);
}

// ============================================================================
// 4) WMMA GEMM: Out(E,M,N) = A(E,M,K)[bf16] x Bw(E,K,N)[bf16]
//    128x128 block tile, K-tile 64, double-buffered LDS,
//    A staged via async global->LDS (ASYNCcnt), B via transpose-scatter.
// ============================================================================
constexpr int TM = 128, TN = 128, TK = 64;
constexpr int LDA_S = 72;   // padded per-row stride of A tile (elements)
constexpr int LDB_S = 72;   // padded per-col stride of transposed B tile

static __device__ __forceinline__ v16bf frag_a(const unsigned short* Alds,
                                               int rbase, int ks, int lane) {
  const int row = rbase + (lane & 15);
  const int kb  = ks + ((lane & 16) ? 8 : 0);
  const unsigned short* p = Alds + row * LDA_S + kb;
  const v8bf lo = *(const v8bf*)p;            // K = kb .. kb+7
  const v8bf hi = *(const v8bf*)(p + 16);     // K = kb+16 .. kb+23
  v16bf a;
#pragma unroll
  for (int i = 0; i < 8; ++i) { a[i] = lo[i]; a[i + 8] = hi[i]; }
  return a;
}

static __device__ __forceinline__ v16bf frag_b(const unsigned short* Blds,
                                               int cbase, int ks, int lane) {
  const int col = cbase + (lane & 15);
  const int kb  = ks + ((lane & 16) ? 8 : 0);
  const unsigned short* p = Blds + col * LDB_S + kb;
  const v8bf lo = *(const v8bf*)p;            // K = kb .. kb+7
  const v8bf hi = *(const v8bf*)(p + 16);     // K = kb+16 .. kb+23
  v16bf b;
#pragma unroll
  for (int i = 0; i < 8; ++i) { b[i] = lo[i]; b[i + 8] = hi[i]; }
  return b;
}

template <bool SILU>
__global__ __launch_bounds__(256) void moe_gemm(
    const unsigned short* __restrict__ A, const unsigned short* __restrict__ Bw,
    void* __restrict__ Out, int M, int N, int K) {
  __shared__ __attribute__((aligned(16))) unsigned short Alds[2][TM * LDA_S];
  __shared__ __attribute__((aligned(16))) unsigned short Blds[2][TN * LDB_S];

  const int e   = blockIdx.z;
  const int m0  = blockIdx.y * TM;
  const int n0  = blockIdx.x * TN;
  const int tid = threadIdx.x;
  const int lane = tid & 31;
  const int wave = tid >> 5;
  const int wr   = wave >> 2;        // 0..1  (64-row group)
  const int wc   = wave & 3;         // 0..3  (32-col group)

  const unsigned short* Ae = A + (size_t)e * M * K;
  const unsigned short* Be = Bw + (size_t)e * K * N;

  v8f acc[4][2];
#pragma unroll
  for (int i = 0; i < 4; ++i)
#pragma unroll
    for (int j = 0; j < 2; ++j) acc[i][j] = (v8f)0.f;

  // loader mapping
  const int a_row = tid >> 1;                 // 0..127
  const int a_col = (tid & 1) * 32;           // 0 or 32 (bf16 elements)
  const int b_row = tid >> 2;                 // 0..63 (K)
  const int b_col = (tid & 3) * 32;           // 0..96 (N)

  uint4 breg[4];                              // 32 bf16 of B

  auto async_a = [&](int kk, int buf) {
    const unsigned short* ag = Ae + (size_t)(m0 + a_row) * K + kk + a_col;
    const unsigned lds_off =
        (unsigned)(unsigned long long)(void*)&Alds[buf][a_row * LDA_S + a_col];
    asm volatile(
        "global_load_async_to_lds_b128 %0, %1, off\n\t"
        "global_load_async_to_lds_b128 %0, %1, off offset:16\n\t"
        "global_load_async_to_lds_b128 %0, %1, off offset:32\n\t"
        "global_load_async_to_lds_b128 %0, %1, off offset:48"
        :: "v"(lds_off), "v"(ag) : "memory");
  };
  auto load_b = [&](int kk) {
    const unsigned short* bg = Be + (size_t)(kk + b_row) * N + n0 + b_col;
#pragma unroll
    for (int i = 0; i < 4; ++i) breg[i] = ((const uint4*)bg)[i];
  };
  auto scatter_b = [&](int buf) {
    const unsigned short* br = (const unsigned short*)breg;
#pragma unroll
    for (int j = 0; j < 32; ++j)
      Blds[buf][(b_col + j) * LDB_S + b_row] = br[j];   // transpose into LDS
  };

  // ---- preload tile 0 into buffer 0 ----
  async_a(0, 0);
  load_b(0);
  scatter_b(0);
  asm volatile("s_wait_asynccnt 0x0" ::: "memory");
  __syncthreads();

  int p = 0;
  for (int k0 = 0; k0 < K; k0 += TK) {
    const int nxt = p ^ 1;
    const bool more = (k0 + TK) < K;
    if (more) {                         // prefetch next tile while computing
      async_a(k0 + TK, nxt);
      load_b(k0 + TK);
    }

    // ---- compute on buffer p: 2 K-steps of 32 ----
#pragma unroll
    for (int ks = 0; ks < TK; ks += 32) {
      v16bf af[4], bf_[2];
#pragma unroll
      for (int i = 0; i < 4; ++i) af[i] = frag_a(Alds[p], wr * 64 + i * 16, ks, lane);
#pragma unroll
      for (int j = 0; j < 2; ++j) bf_[j] = frag_b(Blds[p], wc * 32 + j * 16, ks, lane);
#pragma unroll
      for (int i = 0; i < 4; ++i)
#pragma unroll
        for (int j = 0; j < 2; ++j)
          acc[i][j] = __builtin_amdgcn_wmma_f32_16x16x32_bf16(
              false, af[i], false, bf_[j], (short)0, acc[i][j], false, false);
    }

    if (more) scatter_b(nxt);
    asm volatile("s_wait_asynccnt 0x0" ::: "memory");
    __syncthreads();
    p = nxt;
  }

  // ---- epilogue ----
  const int cl   = lane & 15;
  const int rofs = (lane & 16) ? 8 : 0;
#pragma unroll
  for (int i = 0; i < 4; ++i)
#pragma unroll
    for (int j = 0; j < 2; ++j)
#pragma unroll
      for (int t = 0; t < 8; ++t) {
        const int row = m0 + wr * 64 + i * 16 + rofs + t;
        const int col = n0 + wc * 32 + j * 16 + cl;
        const size_t idx = ((size_t)e * M + row) * N + col;
        const float x = acc[i][j][t];
        if (SILU) {
          const float s = x / (1.f + __expf(-x));          // silu
          ((unsigned short*)Out)[idx] = f2bf_bits(s);
        } else {
          ((float*)Out)[idx] = x;
        }
      }
}

// ============================================================================
// 5) Combine: hidden = prob * (kept ? Xout[slot] : data)
// ============================================================================
__global__ __launch_bounds__(256) void combine_kernel(
    const float* __restrict__ data, const float* __restrict__ xout,
    const float* __restrict__ prob, const int* __restrict__ eidx,
    const int* __restrict__ slotpos, float* __restrict__ hidden) {
  const int tok = blockIdx.x;
  const float p = prob[tok];
  const int pos = slotpos[tok];
  float4 v;
  if (pos >= 0) {
    const int e = eidx[tok];
    const int b = tok / SS;
    const size_t row = (size_t)e * TT + (size_t)b * CAP + pos;
    v = ((const float4*)(xout + row * DD))[threadIdx.x];
  } else {
    v = ((const float4*)(data + (size_t)tok * DD))[threadIdx.x];
  }
  v.x *= p; v.y *= p; v.z *= p; v.w *= p;
  ((float4*)(hidden + (size_t)tok * DD))[threadIdx.x] = v;
}

// ============================================================================
// launch
// ============================================================================
extern "C" void kernel_launch(void* const* d_in, const int* in_sizes, int n_in,
                              void* d_out, int out_size, void* d_ws, size_t ws_size,
                              hipStream_t stream) {
  const float* data   = (const float*)d_in[0];
  const float* gate_w = (const float*)d_in[1];
  const float* w1     = (const float*)d_in[2];
  const float* w2     = (const float*)d_in[3];
  (void)in_sizes; (void)n_in; (void)out_size; (void)ws_size;

  // ---- workspace layout (bytes, 256-aligned blocks) ----
  char* ws = (char*)d_ws;
  int*            eidx = (int*)(ws + 0);                                  // 32 KB
  int*            slot = (int*)(ws + 32768);                              // 32 KB
  float*          prob = (float*)(ws + 65536);                            // 32 KB
  unsigned short* xin  = (unsigned short*)(ws + 98304);                   // 20.97 MB
  unsigned short* w1b  = (unsigned short*)(ws + 98304 + 20971520ull);     // 67.11 MB
  unsigned short* w2b  = (unsigned short*)(ws + 98304 + 20971520ull + 67108864ull);
  unsigned short* h    = (unsigned short*)(ws + 98304 + 20971520ull + 134217728ull); // 83.89 MB
  float*          xout = (float*)(ws + 98304 + 20971520ull + 134217728ull + 83886080ull);

  // ---- output layout: hidden | logits | expert_index ----
  float* hidden = (float*)d_out;
  float* logits = hidden + (size_t)NTOK * DD;
  int*   ei_out = (int*)(logits + (size_t)NTOK * EE);

  // weight pre-conversion (W becomes bf16 and L2-resident for the GEMMs)
  cvt_bf16_kernel<<<(EE * DD * FF) / (256 * 4), 256, 0, stream>>>(w1, w1b);
  cvt_bf16_kernel<<<(EE * FF * DD) / (256 * 4), 256, 0, stream>>>(w2, w2b);

  router_kernel<<<NTOK / 8, 256, 0, stream>>>(data, gate_w, logits, eidx, prob);
  scan_kernel<<<BB, 32, 0, stream>>>(eidx, slot, ei_out);
  dispatch_kernel<<<NTOK, 256, 0, stream>>>(data, eidx, slot, xin);

  moe_gemm<true ><<<dim3(FF / TN, TT / TM, EE), 256, 0, stream>>>(xin, w1b, h,    TT, FF, DD);
  moe_gemm<false><<<dim3(DD / TN, TT / TM, EE), 256, 0, stream>>>(h,   w2b, xout, TT, DD, FF);

  combine_kernel<<<NTOK, 256, 0, stream>>>(data, xout, prob, eidx, slot, hidden);
}
